// LTCCell_17575006175776
// MI455X (gfx1250) — matrix operand order repaired and checked by reading, not compiled, not run on hardware
//
#include <hip/hip_runtime.h>
#include <hip/hip_bf16.h>

#define BDIM 8192
#define INDIM 512
#define HDIM 1024

typedef __attribute__((ext_vector_type(16))) __bf16 v16bf;
typedef __attribute__((ext_vector_type(8)))  float  v8f;
typedef __attribute__((ext_vector_type(4)))  float  f32x4;

union FragU { v16bf v; f32x4 q[2]; };

// A-operand fragment: row-major [rows, K] bf16, per ISA 16-bit A layout:
// lanes 0-15: M=row, K chunks at [k0..k0+7] and [k0+16..k0+23]
// lanes 16-31: K chunks at [k0+8..k0+15] and [k0+24..k0+31]
__device__ __forceinline__ v16bf load_frag_a(const __bf16* base, int row0, int k0,
                                             int ldk, int lane) {
  int r  = row0 + (lane & 15);
  int kb = k0 + ((lane >> 4) << 3);
  const __bf16* p = base + (size_t)r * ldk + kb;
  FragU f;
  f.q[0] = *reinterpret_cast<const f32x4*>(p);
  f.q[1] = *reinterpret_cast<const f32x4*>(p + 16);
  return f.v;
}

// B-operand fragment for C = A * W^T: W row-major [N, K] bf16.
// Same K order as A: lanes 0-15 hold K=0..7,16..23; lanes 16-31 K=8..15,24..31.
__device__ __forceinline__ v16bf load_frag_b(const __bf16* base, int col0, int k0,
                                             int ldk, int lane) {
  int n  = col0 + (lane & 15);
  int kb = k0 + ((lane >> 4) << 3);
  const __bf16* p = base + (size_t)n * ldk + kb;
  FragU f;
  f.q[0] = *reinterpret_cast<const f32x4*>(p);
  f.q[1] = *reinterpret_cast<const f32x4*>(p + 16);
  return f.v;
}

__device__ __forceinline__ v8f wmma_bf16(v16bf a, v16bf b, v8f c) {
  return __builtin_amdgcn_wmma_f32_16x16x32_bf16(false, a, false, b, (short)0, c,
                                                 false, false);
}

struct DualAcc { v8f r[2][2]; v8f p[2][2]; };

// Dual GEMM: R = A @ W1^T tile, P = A @ W2^T tile, shared A operand.
// Wave computes 32x32 output (2x2 WMMA accumulators per GEMM).
__device__ __forceinline__ void dual_gemm(const __bf16* A, const __bf16* W1,
                                          const __bf16* W2, int row0, int col0,
                                          int K, int lane, DualAcc& d) {
  v8f z = {};
#pragma unroll
  for (int i = 0; i < 2; ++i)
#pragma unroll
    for (int j = 0; j < 2; ++j) { d.r[i][j] = z; d.p[i][j] = z; }

  for (int k0 = 0; k0 < K; k0 += 32) {
    v16bf a0 = load_frag_a(A, row0,      k0, K, lane);
    v16bf a1 = load_frag_a(A, row0 + 16, k0, K, lane);
    v16bf b0 = load_frag_b(W1, col0,      k0, K, lane);
    v16bf b1 = load_frag_b(W1, col0 + 16, k0, K, lane);
    v16bf c0 = load_frag_b(W2, col0,      k0, K, lane);
    v16bf c1 = load_frag_b(W2, col0 + 16, k0, K, lane);
    d.r[0][0] = wmma_bf16(a0, b0, d.r[0][0]);
    d.r[0][1] = wmma_bf16(a0, b1, d.r[0][1]);
    d.r[1][0] = wmma_bf16(a1, b0, d.r[1][0]);
    d.r[1][1] = wmma_bf16(a1, b1, d.r[1][1]);
    d.p[0][0] = wmma_bf16(a0, c0, d.p[0][0]);
    d.p[0][1] = wmma_bf16(a0, c1, d.p[0][1]);
    d.p[1][0] = wmma_bf16(a1, c0, d.p[1][0]);
    d.p[1][1] = wmma_bf16(a1, c1, d.p[1][1]);
  }
}

// ---------------- conversion / setup kernels ----------------

__global__ void cvt_bf16(const float* __restrict__ src,
                         __hip_bfloat16* __restrict__ dst, long n) {
  long i = (long)blockIdx.x * blockDim.x + threadIdx.x;
  long stride = (long)gridDim.x * blockDim.x;
  for (; i < n; i += stride) dst[i] = __float2bfloat16(src[i]);
}

__global__ void cvt_split_gate(const float* __restrict__ gw,
                               __hip_bfloat16* __restrict__ gwx,
                               __hip_bfloat16* __restrict__ gwh) {
  const long W = INDIM + HDIM;
  const long n = (long)HDIM * W;
  long i = (long)blockIdx.x * blockDim.x + threadIdx.x;
  long stride = (long)gridDim.x * blockDim.x;
  for (; i < n; i += stride) {
    long row = i / W, col = i % W;
    __hip_bfloat16 v = __float2bfloat16(gw[i]);
    if (col < INDIM) gwx[row * INDIM + col] = v;
    else             gwh[row * HDIM + (col - INDIM)] = v;
  }
}

__global__ void tau_kernel(const float* __restrict__ tau,
                           float* __restrict__ invtau) {
  int i = blockIdx.x * blockDim.x + threadIdx.x;
  if (i < HDIM) {
    float t = tau[i];
    float sp = (t > 20.0f) ? t : log1pf(expf(t));   // softplus
    invtau[i] = 1.0f / (sp + 1.0f);
  }
}

// ---------------- constant-drive dual GEMM: x_in and gate_x ----------------

__global__ __launch_bounds__(256) void gemm_xin_gx(
    const __bf16* __restrict__ xbf, const __bf16* __restrict__ winb,
    const __bf16* __restrict__ gwxb, const float* __restrict__ gate_b,
    float* __restrict__ xin, float* __restrict__ gx) {
  int lane = threadIdx.x & 31;
  int wave = threadIdx.x >> 5;
  int row0 = blockIdx.y * 128 + (wave >> 1) * 32;
  int col0 = blockIdx.x * 64  + (wave & 1) * 32;
  DualAcc d;
  dual_gemm(xbf, winb, gwxb, row0, col0, INDIM, lane, d);
#pragma unroll
  for (int mt = 0; mt < 2; ++mt)
#pragma unroll
    for (int nt = 0; nt < 2; ++nt) {
      int n  = col0 + nt * 16 + (lane & 15);
      int rb = row0 + mt * 16 + ((lane >> 4) << 3);
      float gb = gate_b[n];
#pragma unroll
      for (int e = 0; e < 8; ++e) {
        size_t idx = (size_t)(rb + e) * HDIM + n;
        xin[idx] = d.r[mt][nt][e];
        gx[idx]  = d.p[mt][nt][e] + gb;
      }
    }
}

// ---------------- RK4 stage: dual GEMM + fused elementwise ----------------
// flags: bit0 = accumulate into acc (else overwrite), bit1 = write next hh.

__global__ __launch_bounds__(256) void stage_kernel(
    const __bf16* __restrict__ hbf, const __bf16* __restrict__ wrecb,
    const __bf16* __restrict__ gwhb, const float* __restrict__ xin,
    const float* __restrict__ gx, const float* __restrict__ hcur,
    const float* __restrict__ h0, const float* __restrict__ invtau,
    float* __restrict__ acc, float* __restrict__ hnf,
    __hip_bfloat16* __restrict__ hnb, float wk, float alpha, int flags) {
  int lane = threadIdx.x & 31;
  int wave = threadIdx.x >> 5;
  int row0 = blockIdx.y * 128 + (wave >> 1) * 32;
  int col0 = blockIdx.x * 64  + (wave & 1) * 32;
  DualAcc d;
  dual_gemm(hbf, wrecb, gwhb, row0, col0, HDIM, lane, d);
#pragma unroll
  for (int mt = 0; mt < 2; ++mt)
#pragma unroll
    for (int nt = 0; nt < 2; ++nt) {
      int n  = col0 + nt * 16 + (lane & 15);
      int rb = row0 + mt * 16 + ((lane >> 4) << 3);
      float itau = invtau[n];
#pragma unroll
      for (int e = 0; e < 8; ++e) {
        size_t idx = (size_t)(rb + e) * HDIM + n;
        float pre  = gx[idx] + d.p[mt][nt][e];
        float th   = tanhf(pre);
        float gate = 1.0f / (1.0f + __expf(-th));  // sigmoid(tanh(pre))
        float k = xin[idx] - hcur[idx] * itau + gate * d.r[mt][nt][e];
        float a = (flags & 1) ? (acc[idx] + wk * k) : (wk * k);
        acc[idx] = a;
        if (flags & 2) {
          float hn = h0[idx] + alpha * k;
          hnf[idx] = hn;
          hnb[idx] = __float2bfloat16(hn);
        }
      }
    }
}

// ---------------- final: h + acc/6 -> LayerNorm -> tanh ----------------

__global__ __launch_bounds__(256) void ln_tanh(
    const float* __restrict__ h0, const float* __restrict__ acc,
    const float* __restrict__ g, const float* __restrict__ b,
    float* __restrict__ out) {
  __shared__ float s1[256];
  __shared__ float s2[256];
  int row = blockIdx.x;
  int t = threadIdx.x;
  const float* hr = h0 + (size_t)row * HDIM;
  const float* ar = acc + (size_t)row * HDIM;
  float v[4];
  float s = 0.0f, ss = 0.0f;
#pragma unroll
  for (int i = 0; i < 4; ++i) {
    int c = t + i * 256;
    float hv = hr[c] + ar[c] * (1.0f / 6.0f);
    v[i] = hv; s += hv; ss += hv * hv;
  }
  s1[t] = s; s2[t] = ss;
  __syncthreads();
  for (int off = 128; off > 0; off >>= 1) {
    if (t < off) { s1[t] += s1[t + off]; s2[t] += s2[t + off]; }
    __syncthreads();
  }
  float mu   = s1[0] * (1.0f / HDIM);
  float var  = s2[0] * (1.0f / HDIM) - mu * mu;
  float rinv = rsqrtf(var + 1e-5f);
#pragma unroll
  for (int i = 0; i < 4; ++i) {
    int c = t + i * 256;
    out[(size_t)row * HDIM + c] = tanhf((v[i] - mu) * rinv * g[c] + b[c]);
  }
}

// ---------------- host launcher ----------------

extern "C" void kernel_launch(void* const* d_in, const int* in_sizes, int n_in,
                              void* d_out, int out_size, void* d_ws, size_t ws_size,
                              hipStream_t stream) {
  const float* x      = (const float*)d_in[0];
  const float* h      = (const float*)d_in[1];
  const float* tau    = (const float*)d_in[2];
  const float* W_in   = (const float*)d_in[3];
  const float* W_rec  = (const float*)d_in[4];
  const float* gate_w = (const float*)d_in[5];
  const float* gate_b = (const float*)d_in[6];
  const float* ln_g   = (const float*)d_in[7];
  const float* ln_b   = (const float*)d_in[8];
  float* out = (float*)d_out;

  const size_t B = BDIM, IN = INDIM, H = HDIM;
  char* p = (char*)d_ws;
  auto alloc = [&](size_t bytes) -> void* {
    void* r = (void*)p;
    p += (bytes + 255) & ~(size_t)255;
    return r;
  };
  __hip_bfloat16* xbf   = (__hip_bfloat16*)alloc(B * IN * 2);
  __hip_bfloat16* winb  = (__hip_bfloat16*)alloc(H * IN * 2);
  __hip_bfloat16* wrecb = (__hip_bfloat16*)alloc(H * H * 2);
  __hip_bfloat16* gwxb  = (__hip_bfloat16*)alloc(H * IN * 2);
  __hip_bfloat16* gwhb  = (__hip_bfloat16*)alloc(H * H * 2);
  __hip_bfloat16* hbf0  = (__hip_bfloat16*)alloc(B * H * 2);
  __hip_bfloat16* hbf1  = (__hip_bfloat16*)alloc(B * H * 2);
  float* xin    = (float*)alloc(B * H * 4);
  float* gx     = (float*)alloc(B * H * 4);
  float* acc    = (float*)alloc(B * H * 4);
  float* hf     = (float*)alloc(B * H * 4);
  float* invtau = (float*)alloc(H * 4);

  cvt_bf16<<<2048, 256, 0, stream>>>(x, xbf, (long)(B * IN));
  cvt_bf16<<<512,  256, 0, stream>>>(W_in, winb, (long)(H * IN));
  cvt_bf16<<<1024, 256, 0, stream>>>(W_rec, wrecb, (long)(H * H));
  cvt_split_gate<<<2048, 256, 0, stream>>>(gate_w, gwxb, gwhb);
  cvt_bf16<<<4096, 256, 0, stream>>>(h, hbf0, (long)(B * H));
  tau_kernel<<<4, 256, 0, stream>>>(tau, invtau);

  dim3 grid(H / 64, B / 128);
  gemm_xin_gx<<<grid, 256, 0, stream>>>((const __bf16*)xbf, (const __bf16*)winb,
                                        (const __bf16*)gwxb, gate_b, xin, gx);

  // RK4: k1 (alpha 0.5, w 1), k2 (alpha 0.5, w 2), k3 (alpha 1.0, w 2), k4 (w 1)
  stage_kernel<<<grid, 256, 0, stream>>>((const __bf16*)hbf0, (const __bf16*)wrecb,
      (const __bf16*)gwhb, xin, gx, h, h, invtau, acc, hf, hbf1, 1.0f, 0.5f, 2);
  stage_kernel<<<grid, 256, 0, stream>>>((const __bf16*)hbf1, (const __bf16*)wrecb,
      (const __bf16*)gwhb, xin, gx, hf, h, invtau, acc, hf, hbf0, 2.0f, 0.5f, 3);
  stage_kernel<<<grid, 256, 0, stream>>>((const __bf16*)hbf0, (const __bf16*)wrecb,
      (const __bf16*)gwhb, xin, gx, hf, h, invtau, acc, hf, hbf1, 2.0f, 1.0f, 3);
  stage_kernel<<<grid, 256, 0, stream>>>((const __bf16*)hbf1, (const __bf16*)wrecb,
      (const __bf16*)gwhb, xin, gx, hf, h, invtau, acc, hf, hbf0, 1.0f, 0.0f, 1);

  ln_tanh<<<B, 256, 0, stream>>>(h, acc, ln_g, ln_b, out);
}
